// NCECriterion_1657857376324
// MI455X (gfx1250) — matrix operand order repaired and checked by reading, not compiled, not run on hardware
//
#include <hip/hip_runtime.h>

// ---------------------------------------------------------------------------
// NCE criterion: loss = -(1/B) * sum_b log( sum_pos exp(s) / sum_all exp(s) )
// s = (x @ feat^T) / (N*T),  B=512, N=100000, D=128, N*T=7000.
// Z1 normalizer in the reference cancels exactly in the ratio.
// GEMM runs on v_wmma_f32_16x16x32_bf16 (f32 accum); feat converted f32->bf16
// via native __bf16 casts (v_cvt_pk_bf16_f32), staged through double-buffered
// LDS (1 barrier/tile, next tile's global loads overlap the WMMAs);
// per-sample sums via shfl_xor + f32 global atomics into d_ws.
// ---------------------------------------------------------------------------

typedef __attribute__((ext_vector_type(16))) __bf16 v16bf;
typedef __attribute__((ext_vector_type(8)))  __bf16 v8bf;
typedef __attribute__((ext_vector_type(8)))  float  v8f;

constexpr int   BN         = 512;            // batch
constexpr int   NN         = 100000;         // memory bank rows
constexpr int   DD         = 128;            // feature dim
constexpr int   NTILES     = NN / 16;        // 6250 column tiles
constexpr int   NCHUNK     = 125;            // -> 50 tiles per chunk exactly
constexpr int   ROWGRP     = 4;              // 512 rows / (8 waves * 16 rows)
constexpr int   LDS_STRIDE = 136;            // halves per feat row (68 dwords)
// exp(s/7000) computed as exp2(s * (log2(e)/7000)) -> single v_exp_f32
constexpr float SCALE_L2E  = 1.4426950408889634f / 7000.0f;

union FragB { v16bf v; uint4 q[2]; };
union AccC  { v8f v; float f[8]; };

__global__ void nce_zero(float* __restrict__ p, int n) {
  int i = blockIdx.x * blockDim.x + threadIdx.x;
  if (i < n) p[i] = 0.0f;
}

__global__ __launch_bounds__(256)
void nce_main(const float* __restrict__ x, const float* __restrict__ feat,
              const int* __restrict__ labels, const int* __restrict__ labf,
              float* __restrict__ gAll, float* __restrict__ gPos) {
  __shared__ __align__(16) unsigned short sFeat[2][16 * LDS_STRIDE];

  const int tid  = threadIdx.x;
  const int lane = tid & 31;
  const int wave = tid >> 5;
  const int hi   = lane >> 4;        // 0: lanes 0-15, 1: lanes 16-31
  const int l15  = lane & 15;
  const int m0   = blockIdx.x * 128 + wave * 16;   // this wave's 16 sample rows

  // staging coords: 256 threads cover a 16x128 f32 tile, 8 elems each
  const int srow = tid >> 4;
  const int scol = (tid & 15) * 8;

  // ---- A fragments: x[m0..m0+15][0..127] as bf16, per 16-bit A 16x32 layout.
  // lanes<16 hold K = {32f+0..7, 32f+16..23}; lanes>=16 hold K = {+8..15, +24..31}
  v16bf a[4];
  {
    const float* xr = x + (size_t)(m0 + l15) * DD;
    const int off = hi * 8;
#pragma unroll
    for (int f = 0; f < 4; ++f) {
      const int k1 = 32 * f + off;
      float4 p0 = *(const float4*)(xr + k1);
      float4 p1 = *(const float4*)(xr + k1 + 4);
      float4 p2 = *(const float4*)(xr + k1 + 16);
      float4 p3 = *(const float4*)(xr + k1 + 20);
      v16bf av;
      av[0]  = (__bf16)p0.x; av[1]  = (__bf16)p0.y;
      av[2]  = (__bf16)p0.z; av[3]  = (__bf16)p0.w;
      av[4]  = (__bf16)p1.x; av[5]  = (__bf16)p1.y;
      av[6]  = (__bf16)p1.z; av[7]  = (__bf16)p1.w;
      av[8]  = (__bf16)p2.x; av[9]  = (__bf16)p2.y;
      av[10] = (__bf16)p2.z; av[11] = (__bf16)p2.w;
      av[12] = (__bf16)p3.x; av[13] = (__bf16)p3.y;
      av[14] = (__bf16)p3.z; av[15] = (__bf16)p3.w;
      a[f] = av;
    }
  }

  // labels for the 8 sample rows this lane accumulates (C layout: M = r + 8*hi)
  int lab[8];
#pragma unroll
  for (int r = 0; r < 8; ++r) lab[r] = labels[m0 + r + hi * 8];

  float accAll[8], accPos[8];
#pragma unroll
  for (int r = 0; r < 8; ++r) { accAll[r] = 0.0f; accPos[r] = 0.0f; }

  const int tpc = NTILES / NCHUNK;            // 50, exact
  const int t0  = blockIdx.y * tpc;
  const int t1  = t0 + tpc;

  // ---- prologue: stage tile t0 into buffer 0
  {
    const float* fr = feat + (size_t)(t0 * 16 + srow) * DD + scol;
    float4 q0 = *(const float4*)fr;
    float4 q1 = *(const float4*)(fr + 4);
    v8bf h;
    h[0] = (__bf16)q0.x; h[1] = (__bf16)q0.y;
    h[2] = (__bf16)q0.z; h[3] = (__bf16)q0.w;
    h[4] = (__bf16)q1.x; h[5] = (__bf16)q1.y;
    h[6] = (__bf16)q1.z; h[7] = (__bf16)q1.w;
    *(v8bf*)&sFeat[0][srow * LDS_STRIDE + scol] = h;
  }

  int p = 0;
  for (int t = t0; t < t1; ++t) {
    __syncthreads();   // buf[p] staged; buf[p^1] no longer being read

    // ---- issue next tile's global loads early (in flight under the WMMAs)
    const bool more = (t + 1 < t1);            // block-uniform
    float4 q0, q1;
    if (more) {
      const float* fr = feat + (size_t)((t + 1) * 16 + srow) * DD + scol;
      q0 = *(const float4*)fr;
      q1 = *(const float4*)(fr + 4);
    }
    // this tile's feat label: one cacheline, hits WGP$/L2
    const int lf = labf[t * 16 + l15];

    // ---- 16x16 tile: 4 chained bf16 WMMAs over K=128 (EXEC all ones)
    AccC c;
#pragma unroll
    for (int r = 0; r < 8; ++r) c.f[r] = 0.0f;
#pragma unroll
    for (int f = 0; f < 4; ++f) {
      // B 32x16 layout: lanes<16 hold K=32f+{0..7,16..23} (col N=lane),
      //                 lanes>=16 hold K=32f+{8..15,24..31} (col N=lane-16)
      FragB b;
      const unsigned short* src = &sFeat[p][l15 * LDS_STRIDE + 32 * f + hi * 8];
      b.q[0] = *(const uint4*)src;
      b.q[1] = *(const uint4*)(src + 16);
      c.v = __builtin_amdgcn_wmma_f32_16x16x32_bf16(
          /*neg_a=*/false, a[f], /*neg_b=*/false, b.v,
          /*c_mod=*/(short)0, c.v, /*reuse_a=*/false, /*reuse_b=*/false);
    }

    // ---- exp + masked accumulation (lane owns column n = t*16 + l15)
#pragma unroll
    for (int r = 0; r < 8; ++r) {
      // exp(s/7000) == exp2(s * log2(e)/7000): one v_exp_f32, no extra mul
      float e = __builtin_amdgcn_exp2f(c.f[r] * SCALE_L2E);
      accAll[r] += e;
      accPos[r] += (lab[r] == lf) ? e : 0.0f;
    }

    // ---- convert + store next tile into the other buffer
    if (more) {
      v8bf h;
      h[0] = (__bf16)q0.x; h[1] = (__bf16)q0.y;
      h[2] = (__bf16)q0.z; h[3] = (__bf16)q0.w;
      h[4] = (__bf16)q1.x; h[5] = (__bf16)q1.y;
      h[6] = (__bf16)q1.z; h[7] = (__bf16)q1.w;
      *(v8bf*)&sFeat[p ^ 1][srow * LDS_STRIDE + scol] = h;
    }
    p ^= 1;
  }

  // ---- reduce over the 16 lanes sharing each sample row, then global atomics
#pragma unroll
  for (int r = 0; r < 8; ++r) {
    float va = accAll[r], vp = accPos[r];
#pragma unroll
    for (int mm = 1; mm < 16; mm <<= 1) {
      va += __shfl_xor(va, mm, 32);
      vp += __shfl_xor(vp, mm, 32);
    }
    if (l15 == 0) {
      const int m = m0 + r + hi * 8;
      atomicAdd(&gAll[m], va);
      atomicAdd(&gPos[m], vp);
    }
  }
}

__global__ __launch_bounds__(256)
void nce_final(const float* __restrict__ gAll, const float* __restrict__ gPos,
               float* __restrict__ out) {
  __shared__ float red[256];
  const int tid = threadIdx.x;
  float s = 0.0f;
  for (int b = tid; b < BN; b += 256)
    s += logf(gPos[b]) - logf(gAll[b]);
  red[tid] = s;
  __syncthreads();
  for (int k = 128; k > 0; k >>= 1) {
    if (tid < k) red[tid] += red[tid + k];
    __syncthreads();
  }
  if (tid == 0) out[0] = -red[0] / (float)BN;
}

extern "C" void kernel_launch(void* const* d_in, const int* in_sizes, int n_in,
                              void* d_out, int out_size, void* d_ws, size_t ws_size,
                              hipStream_t stream) {
  const float* x      = (const float*)d_in[0];
  const float* feat   = (const float*)d_in[1];
  const int*   labels = (const int*)d_in[2];
  const int*   labf   = (const int*)d_in[3];

  float* gAll = (float*)d_ws;        // [512]
  float* gPos = gAll + BN;           // [512]

  nce_zero<<<dim3((2 * BN + 255) / 256), 256, 0, stream>>>(gAll, 2 * BN);

  dim3 grid(ROWGRP, NCHUNK);         // x fastest: 4 row-groups of a chunk adjacent in L2
  nce_main<<<grid, 256, 0, stream>>>(x, feat, labels, labf, gAll, gPos);

  nce_final<<<1, 256, 0, stream>>>(gAll, gPos, (float*)d_out);
}
